// ContinuousThoughtMachine_26903675142249
// MI455X (gfx1250) — matrix operand order repaired and checked by reading, not compiled, not run on hardware
//
#include <hip/hip_runtime.h>
#include <hip/hip_bf16.h>
#include <cstdint>
#include <cstddef>

// ---------------- CDNA5 WMMA types ----------------
typedef __attribute__((ext_vector_type(16))) __bf16 v16bf;
typedef __attribute__((ext_vector_type(8)))  float  v8f;

union Frag16 { v16bf v; uint4 q[2]; unsigned short u[16]; };

__device__ __forceinline__ unsigned short f2bf(float f) {
  unsigned int u = __float_as_uint(f);
  u += ((u >> 16) & 1u) + 0x7fffu;           // round-to-nearest-even
  return (unsigned short)(u >> 16);
}
__device__ __forceinline__ void cvt4(unsigned short* d, float4 f) {
  d[0] = f2bf(f.x); d[1] = f2bf(f.y); d[2] = f2bf(f.z); d[3] = f2bf(f.w);
}

// ---------------- model constants ----------------
enum { Bz = 16, Ez = 512, Dz = 2048, Mz = 16, Hz = 8, HDz = 64, OUTz = 1000,
       Sz = 256, NSAz = 256, NSOz = 128, REPAz = 32896, REPOz = 8256,
       KNLM = 32768 };

// =========================================================================
// Generic bf16 WMMA GEMM:  C[M x N] = A[M x K](f32,row-major) * Bt[K x N](bf16)
// One wave per block; each wave owns a 16 x (16*NT) tile, K split over grid.z.
// bias == nullptr : accumulate into pre-initialized C via global f32 atomics.
// bias != nullptr : single K-chunk mode, plain stores of bias[n] + acc.
// Inner loop batches all NT B-fragments per k-slice so loads of the next
// slice can be issued while the current slice's WMMAs execute.
// =========================================================================
template <int NT>
__global__ void k_gemm(const float* __restrict__ A, int lda,
                       const unsigned short* __restrict__ Bt, int ldb,
                       float* __restrict__ C, int ldc,
                       int K, int kchunk, const float* __restrict__ bias) {
  int lane = threadIdx.x;                  // wave32
  int n0 = blockIdx.x * (16 * NT);
  int m0 = blockIdx.y * 16;
  int k0 = blockIdx.z * kchunk;
  int kend = (K < k0 + kchunk) ? K : k0 + kchunk;
  int half = lane >> 4;
  int arow = lane & 15;
  const float* Ap = A + (size_t)(m0 + arow) * lda + half * 8;

  v8f zero = {};
  v8f acc[NT];
#pragma unroll
  for (int t = 0; t < NT; ++t) acc[t] = zero;

#pragma unroll 2
  for (int k = k0; k < kend; k += 32) {
    // ---- A fragment (fp32 -> bf16 in registers) ----
    Frag16 a;
    const float* ap = Ap + k;
    cvt4(&a.u[0],  *(const float4*)(ap));        // K = k + half*8 + 0..3
    cvt4(&a.u[4],  *(const float4*)(ap + 4));    //     ... + 4..7
    cvt4(&a.u[8],  *(const float4*)(ap + 16));   // K = k + 16 + half*8 + 0..3
    cvt4(&a.u[12], *(const float4*)(ap + 20));

    // ---- batch all NT B fragments of this k-slice ----
    const unsigned short* bp = Bt + (size_t)(k + half * 8) * ldb + n0 + arow;
    if (k + 64 < kend) __builtin_prefetch(bp + (size_t)64 * ldb, 0, 0);
    Frag16 bf[NT];
#pragma unroll
    for (int t = 0; t < NT; ++t) {
#pragma unroll
      for (int s = 0; s < 16; ++s)
        bf[t].u[s] = bp[(size_t)(s < 8 ? s : s + 8) * ldb + t * 16];
    }
    // ---- NT independent WMMAs ----
#pragma unroll
    for (int t = 0; t < NT; ++t)
      acc[t] = __builtin_amdgcn_wmma_f32_16x16x32_bf16(
          false, a.v, false, bf[t].v, (short)0, acc[t], false, false);
  }

  if (bias != nullptr) {          // single-chunk: direct store
#pragma unroll
    for (int t = 0; t < NT; ++t) {
      float bn = bias[n0 + t * 16 + arow];
#pragma unroll
      for (int r = 0; r < 8; ++r)
        C[(size_t)(m0 + half * 8 + r) * ldc + n0 + t * 16 + arow] = bn + acc[t][r];
    }
  } else {                        // K-split: atomic accumulate
#pragma unroll
    for (int t = 0; t < NT; ++t)
#pragma unroll
      for (int r = 0; r < 8; ++r)
        atomicAdd(&C[(size_t)(m0 + half * 8 + r) * ldc + n0 + t * 16 + arow],
                  acc[t][r]);
  }
}

// =========================================================================
// conv2 implicit GEMM: M=16*32*32 pixels, N=512, K=9*512.
// A read directly from padded NHWC bf16 activations h1p[16][34][34][512].
// =========================================================================
template <int NT>
__global__ void k_conv2gemm(const unsigned short* __restrict__ h1p,
                            const unsigned short* __restrict__ W2T,
                            float* __restrict__ C, int kchunk) {
  int lane = threadIdx.x;
  int n0 = blockIdx.x * (16 * NT);
  int m0 = blockIdx.y * 16;
  int k0 = blockIdx.z * kchunk;
  int kend = (4608 < k0 + kchunk) ? 4608 : k0 + kchunk;
  int half = lane >> 4;
  int arow = lane & 15;
  int b = m0 >> 10, rem = m0 & 1023, yo = rem >> 5, x0 = rem & 31;
  int xo = x0 + arow;

  v8f zero = {};
  v8f acc[NT];
#pragma unroll
  for (int t = 0; t < NT; ++t) acc[t] = zero;

#pragma unroll 2
  for (int k = k0; k < kend; k += 32) {
    int pos = k >> 9, cc = k & 511;
    int kh = pos / 3, kw = pos % 3;
    const unsigned short* ap =
        h1p + (((size_t)(b * 34 + yo + kh) * 34) + (xo + kw)) * 512 + cc + half * 8;
    Frag16 a;
    a.q[0] = *(const uint4*)(ap);
    a.q[1] = *(const uint4*)(ap + 16);

    const unsigned short* bp = W2T + (size_t)(k + half * 8) * 512 + n0 + arow;
    if (k + 64 < kend) __builtin_prefetch(bp + (size_t)64 * 512, 0, 0);
    Frag16 bf[NT];
#pragma unroll
    for (int t = 0; t < NT; ++t) {
#pragma unroll
      for (int s = 0; s < 16; ++s)
        bf[t].u[s] = bp[(size_t)(s < 8 ? s : s + 8) * 512 + t * 16];
    }
#pragma unroll
    for (int t = 0; t < NT; ++t)
      acc[t] = __builtin_amdgcn_wmma_f32_16x16x32_bf16(
          false, a.v, false, bf[t].v, (short)0, acc[t], false, false);
  }
#pragma unroll
  for (int t = 0; t < NT; ++t)
#pragma unroll
    for (int r = 0; r < 8; ++r)
      atomicAdd(&C[(size_t)(m0 + half * 8 + r) * 512 + n0 + t * 16 + arow],
                acc[t][r]);
}

// ---------------- utility kernels ----------------
__global__ void k_fzero(float* p, int n) {
  int t = blockIdx.x * 256 + threadIdx.x;
  if (t < n) p[t] = 0.f;
}
__global__ void k_hzero(unsigned short* p, int n) {
  int t = blockIdx.x * 256 + threadIdx.x;
  if (t < n) p[t] = 0;
}
__global__ void k_rowbias(float* C, const float* __restrict__ bias,
                          int M, int N, int ldc, int Nvalid) {
  int t = blockIdx.x * 256 + threadIdx.x;
  if (t >= M * N) return;
  int m = t / N, n = t % N;
  C[(size_t)m * ldc + n] = (n < Nvalid) ? bias[n] : 0.f;
}
__global__ void k_copy(float* __restrict__ dst, const float* __restrict__ src, int n) {
  int t = blockIdx.x * 256 + threadIdx.x;
  if (t < n) dst[t] = src[t];
}

// tiled transpose: dst[k][n] (bf16, Npad cols) = src[n+rowoff][k] (fp32, K cols)
__global__ void k_tr(unsigned short* __restrict__ dst, const float* __restrict__ src,
                     int N, int K, int Npad, int rowoff) {
  __shared__ float tile[32][33];
  int kb = blockIdx.x * 32, nb = blockIdx.y * 32;
  int tx = threadIdx.x & 31, ty = threadIdx.x >> 5;
  for (int i = ty; i < 32; i += 8) {
    int n = nb + i, k = kb + tx;
    tile[i][tx] = (n < N && k < K) ? src[(size_t)(n + rowoff) * K + k] : 0.f;
  }
  __syncthreads();
  for (int i = ty; i < 32; i += 8) {
    int k = kb + i, n = nb + tx;
    if (k < K && n < Npad) dst[(size_t)k * Npad + n] = f2bf(tile[tx][i]);
  }
}

// conv2 weight: W2T[(kh*3+kw)*512 + ci][co] = conv2_w[co][ci][kh][kw]
__global__ void k_conv2wt(unsigned short* __restrict__ dst,
                          const float* __restrict__ w) {
  int t = blockIdx.x * 256 + threadIdx.x;
  if (t >= 4608 * 512) return;
  int row = t / 512, co = t % 512;
  int pos = row >> 9, ci = row & 511;
  int kh = pos / 3, kw = pos % 3;
  dst[t] = f2bf(w[(((size_t)co * 512 + ci) * 3 + kh) * 3 + kw]);
}

// fused NLM weight: W1fT[(d*16+m)][h] = w1[m][h][d] * w2[h][d]  (tiled transpose)
__global__ void k_nlmfuse(unsigned short* __restrict__ dst,
                          const float* __restrict__ w1,
                          const float* __restrict__ w2) {
  __shared__ float tile[32][33];
  int m = blockIdx.z;
  int db = blockIdx.x * 32, hb = blockIdx.y * 32;
  int tx = threadIdx.x & 31, ty = threadIdx.x >> 5;
  for (int i = ty; i < 32; i += 8) {
    size_t idx = (size_t)(hb + i) * 2048 + (db + tx);
    tile[i][tx] = w1[(size_t)m * 2048 * 2048 + idx] * w2[idx];
  }
  __syncthreads();
  for (int i = ty; i < 32; i += 8)
    dst[((size_t)(db + i) * 16 + m) * 2048 + hb + tx] = f2bf(tile[tx][i]);
}

// c[h] = sum_d b1[h][d]*w2[h][d] + b2[h]
__global__ void k_nlmbias(const float* __restrict__ b1, const float* __restrict__ w2,
                          const float* __restrict__ b2, float* __restrict__ c) {
  int h = blockIdx.x, tid = threadIdx.x;
  __shared__ float red[256];
  float s = 0.f;
  for (int d = tid; d < 2048; d += 256)
    s += b1[(size_t)h * 2048 + d] * w2[(size_t)h * 2048 + d];
  red[tid] = s; __syncthreads();
  for (int k = 128; k > 0; k >>= 1) { if (tid < k) red[tid] += red[tid + k]; __syncthreads(); }
  if (tid == 0) c[h] = red[0] + b2[h];
}

// triu pair index tables + decay factors
__global__ void k_pairidx(unsigned short* I, unsigned short* J, float* r,
                          const float* __restrict__ decay, int NS, int npairs) {
  int p = blockIdx.x * 256 + threadIdx.x;
  if (p >= npairs) return;
  int idx = p, i = 0, run = NS;
  while (idx >= run) { idx -= run; run--; i++; }
  I[p] = (unsigned short)i;
  J[p] = (unsigned short)(i + idx);
  r[p] = __expf(-decay[p]);
}

// pair product + exponential decay accumulators + normalized output
__global__ void k_pair(const float* __restrict__ post, int postoff,
                       const unsigned short* __restrict__ I,
                       const unsigned short* __restrict__ J,
                       const float* __restrict__ r,
                       float* __restrict__ a, float* __restrict__ bv,
                       float* __restrict__ sa, int npairs) {
  int t = blockIdx.x * 256 + threadIdx.x;
  if (t >= 16 * npairs) return;
  int b = t / npairs, p = t % npairs;
  const float* pb = post + (size_t)b * Dz + postoff;
  float pair = pb[I[p]] * pb[J[p]];
  float rr = r[p];
  float av = rr * a[t] + pair;  a[t] = av;
  float bb = rr * bv[t] + 1.f;  bv[t] = bb;
  sa[t] = av * rsqrtf(bb);
}

// layernorm over last dim (one block per row)
__global__ void k_layernorm(const float* __restrict__ X, float* __restrict__ Y,
                            const float* __restrict__ g, const float* __restrict__ b,
                            int dim) {
  int row = blockIdx.x, tid = threadIdx.x;
  const float* x = X + (size_t)row * dim;
  float* y = Y + (size_t)row * dim;
  __shared__ float red[256];
  float s = 0.f;
  for (int i = tid; i < dim; i += 256) s += x[i];
  red[tid] = s; __syncthreads();
  for (int k = 128; k > 0; k >>= 1) { if (tid < k) red[tid] += red[tid + k]; __syncthreads(); }
  float mu = red[0] / dim; __syncthreads();
  float v = 0.f;
  for (int i = tid; i < dim; i += 256) { float d0 = x[i] - mu; v += d0 * d0; }
  red[tid] = v; __syncthreads();
  for (int k = 128; k > 0; k >>= 1) { if (tid < k) red[tid] += red[tid + k]; __syncthreads(); }
  float rstd = rsqrtf(red[0] / dim + 1e-5f);
  for (int i = tid; i < dim; i += 256) y[i] = (x[i] - mu) * rstd * g[i] + b[i];
}

// conv1 + bn1 + relu + 2x2 maxpool, output padded NHWC bf16 [16][34][34][512]
__global__ void k_conv1(const float* __restrict__ x, const float* __restrict__ w,
                        const float* __restrict__ cb, const float* __restrict__ g,
                        const float* __restrict__ bb, const float* __restrict__ bm,
                        const float* __restrict__ bvv,
                        unsigned short* __restrict__ h1p) {
  int t = blockIdx.x * 256 + threadIdx.x;
  if (t >= 16 * 32 * 32 * 512) return;
  int co = t & 511; int r = t >> 9;
  int xo = r & 31; r >>= 5;
  int yo = r & 31; int b = r >> 5;
  float scale = g[co] * rsqrtf(bvv[co] + 1e-5f);
  float shift = bb[co] - bm[co] * scale;
  float best = -3.4e38f;
  for (int py = 0; py < 2; ++py)
    for (int px = 0; px < 2; ++px) {
      int yi0 = yo * 2 + py, xi0 = xo * 2 + px;
      float acc = cb[co];
      for (int ci = 0; ci < 3; ++ci)
        for (int kh = 0; kh < 3; ++kh) {
          int yy = yi0 + kh - 1;
          if (yy < 0 || yy >= 64) continue;
          for (int kw = 0; kw < 3; ++kw) {
            int xx = xi0 + kw - 1;
            if (xx < 0 || xx >= 64) continue;
            acc += x[(((size_t)b * 3 + ci) * 64 + yy) * 64 + xx] *
                   w[(((size_t)co * 3 + ci) * 3 + kh) * 3 + kw];
          }
        }
      float v = scale * acc + shift;
      v = v > 0.f ? v : 0.f;
      best = fmaxf(best, v);
    }
  h1p[(((size_t)b * 34 + yo + 1) * 34 + xo + 1) * 512 + co] = f2bf(best);
}

// bn2 + relu + 2x2 maxpool -> feat[b][s][c]
__global__ void k_pool2feat(const float* __restrict__ craw,
                            const float* __restrict__ g, const float* __restrict__ bb,
                            const float* __restrict__ bm, const float* __restrict__ bvv,
                            float* __restrict__ feat) {
  int t = blockIdx.x * 256 + threadIdx.x;
  if (t >= 16 * 16 * 16 * 512) return;
  int c = t & 511; int r = t >> 9;
  int xo = r & 15; r >>= 4;
  int yo = r & 15; int b = r >> 4;
  float scale = g[c] * rsqrtf(bvv[c] + 1e-5f);
  float shift = bb[c] - bm[c] * scale;
  float best = -3.4e38f;
  for (int py = 0; py < 2; ++py)
    for (int px = 0; px < 2; ++px) {
      float v = craw[(((size_t)b * 32 + yo * 2 + py) * 32 + xo * 2 + px) * 512 + c];
      v = scale * v + shift;
      v = v > 0.f ? v : 0.f;
      best = fmaxf(best, v);
    }
  feat[((size_t)b * 256 + yo * 16 + xo) * 512 + c] = best;
}

// attention for one (b,h): softmax(q.K/8).V
__global__ void k_attn(const float* __restrict__ qh, const float* __restrict__ Kb,
                       const float* __restrict__ Vb, float* __restrict__ o) {
  int bh = blockIdx.x;
  int b = bh >> 3, h = bh & 7;
  int tid = threadIdx.x;
  __shared__ float qs[64];
  __shared__ float att[256];
  __shared__ float red[256];
  if (tid < 64) qs[tid] = qh[(size_t)b * 512 + h * 64 + tid];
  __syncthreads();
  const float* krow = Kb + ((size_t)(b * 256 + tid)) * 512 + h * 64;
  float sc = 0.f;
  for (int d = 0; d < 64; ++d) sc += qs[d] * krow[d];
  sc *= 0.125f;
  red[tid] = sc; __syncthreads();
  for (int s = 128; s > 0; s >>= 1) { if (tid < s) red[tid] = fmaxf(red[tid], red[tid + s]); __syncthreads(); }
  float mx = red[0]; __syncthreads();
  float e = __expf(sc - mx);
  red[tid] = e; __syncthreads();
  for (int s = 128; s > 0; s >>= 1) { if (tid < s) red[tid] += red[tid + s]; __syncthreads(); }
  float inv = 1.f / red[0];
  att[tid] = e * inv;
  __syncthreads();
  int d = tid & 63, grp = tid >> 6;
  float acc = 0.f;
  for (int s = grp; s < 256; s += 4)
    acc += att[s] * Vb[((size_t)(b * 256 + s)) * 512 + h * 64 + d];
  red[tid] = acc; __syncthreads();
  if (grp == 0)
    o[(size_t)b * 512 + h * 64 + d] = red[d] + red[64 + d] + red[128 + d] + red[192 + d];
}

// zin = [oproj(512), post(2048)]
__global__ void k_zin(float* __restrict__ zin, const float* __restrict__ oproj,
                      const float* __restrict__ post) {
  int t = blockIdx.x * 256 + threadIdx.x;
  if (t >= 16 * 2560) return;
  int b = t / 2560, c = t % 2560;
  zin[t] = (c < 512) ? oproj[(size_t)b * 512 + c] : post[(size_t)b * 2048 + (c - 512)];
}

// glu: preraw = z[:, :D] * sigmoid(z[:, D:])
__global__ void k_glu(float* __restrict__ preraw, const float* __restrict__ z) {
  int t = blockIdx.x * 256 + threadIdx.x;
  if (t >= 16 * 2048) return;
  int b = t >> 11, d = t & 2047;
  float zb = z[(size_t)b * 4096 + 2048 + d];
  preraw[t] = z[(size_t)b * 4096 + d] * (1.f / (1.f + __expf(-zb)));
}

// hist[b][d][0..14] <- hist[b][d][1..15]; hist[b][d][15] <- pre[b][d]
__global__ void k_histshift(float* __restrict__ hist, const float* __restrict__ pre) {
  int t = blockIdx.x * 256 + threadIdx.x;
  if (t >= 16 * 2048) return;
  size_t base = (size_t)t * 16;
#pragma unroll
  for (int m = 0; m < 15; ++m) hist[base + m] = hist[base + m + 1];
  hist[base + 15] = pre[t];
}

__global__ void k_histinit(float* __restrict__ hist, const float* __restrict__ h0) {
  int t = blockIdx.x * 256 + threadIdx.x;
  if (t >= 16 * 2048 * 16) return;
  hist[t] = h0[t & (2048 * 16 - 1)];
}
__global__ void k_postinit(float* __restrict__ post, const float* __restrict__ p0) {
  int t = blockIdx.x * 256 + threadIdx.x;
  if (t >= 16 * 2048) return;
  post[t] = p0[t & 2047];
}

// pred write-out + normalized entropy certainty
__global__ void k_softout(const float* __restrict__ C, float* __restrict__ pred,
                          float* __restrict__ cert, int step) {
  int b = blockIdx.x, tid = threadIdx.x;
  __shared__ float red[256];
  float mx = -3.4e38f;
  for (int o = tid; o < 1000; o += 256) mx = fmaxf(mx, C[(size_t)b * 1008 + o]);
  red[tid] = mx; __syncthreads();
  for (int s = 128; s > 0; s >>= 1) { if (tid < s) red[tid] = fmaxf(red[tid], red[tid + s]); __syncthreads(); }
  mx = red[0]; __syncthreads();
  float se = 0.f, sxe = 0.f;
  for (int o = tid; o < 1000; o += 256) {
    float xv = C[(size_t)b * 1008 + o] - mx;
    float e = __expf(xv);
    se += e; sxe += xv * e;
  }
  red[tid] = se; __syncthreads();
  for (int s = 128; s > 0; s >>= 1) { if (tid < s) red[tid] += red[tid + s]; __syncthreads(); }
  float Z = red[0]; __syncthreads();
  red[tid] = sxe; __syncthreads();
  for (int s = 128; s > 0; s >>= 1) { if (tid < s) red[tid] += red[tid + s]; __syncthreads(); }
  float sxeT = red[0];
  float ne = (logf(Z) - sxeT / Z) / logf(1000.f);
  for (int o = tid; o < 1000; o += 256)
    pred[((size_t)b * 1000 + o) * 16 + step] = C[(size_t)b * 1008 + o];
  if (tid == 0) {
    cert[b * 32 + step] = ne;
    cert[b * 32 + 16 + step] = 1.f - ne;
  }
}

// =========================================================================
// Host driver
// =========================================================================
static inline int cdiv(int a, int b) { return (a + b - 1) / b; }

extern "C" void kernel_launch(void* const* d_in, const int* in_sizes, int n_in,
                              void* d_out, int out_size, void* d_ws, size_t ws_size,
                              hipStream_t stream) {
  (void)in_sizes; (void)n_in; (void)out_size; (void)ws_size;
  const float* in_x      = (const float*)d_in[0];
  const float* conv1_w   = (const float*)d_in[1];
  const float* conv1_b   = (const float*)d_in[2];
  const float* bn1_g     = (const float*)d_in[3];
  const float* bn1_b     = (const float*)d_in[4];
  const float* bn1_m     = (const float*)d_in[5];
  const float* bn1_v     = (const float*)d_in[6];
  const float* conv2_w   = (const float*)d_in[7];
  const float* conv2_b   = (const float*)d_in[8];
  const float* bn2_g     = (const float*)d_in[9];
  const float* bn2_b     = (const float*)d_in[10];
  const float* bn2_m     = (const float*)d_in[11];
  const float* bn2_v     = (const float*)d_in[12];
  const float* kv_w      = (const float*)d_in[13];
  const float* kv_b      = (const float*)d_in[14];
  const float* kv_ln_g   = (const float*)d_in[15];
  const float* kv_ln_b   = (const float*)d_in[16];
  const float* q_w       = (const float*)d_in[17];
  const float* q_b       = (const float*)d_in[18];
  const float* q_ln_g    = (const float*)d_in[19];
  const float* q_ln_b    = (const float*)d_in[20];
  const float* attn_in_w = (const float*)d_in[21];
  const float* attn_in_b = (const float*)d_in[22];
  const float* attn_out_w= (const float*)d_in[23];
  const float* attn_out_b= (const float*)d_in[24];
  const float* syn_w     = (const float*)d_in[25];
  const float* syn_b     = (const float*)d_in[26];
  const float* syn_ln_g  = (const float*)d_in[27];
  const float* syn_ln_b  = (const float*)d_in[28];
  const float* nlm_w1    = (const float*)d_in[29];
  const float* nlm_b1    = (const float*)d_in[30];
  const float* nlm_w2    = (const float*)d_in[31];
  const float* nlm_b2    = (const float*)d_in[32];
  const float* post_init = (const float*)d_in[33];
  const float* hist_init = (const float*)d_in[34];
  const float* decay_a   = (const float*)d_in[35];
  const float* decay_o   = (const float*)d_in[36];
  const float* out_w     = (const float*)d_in[37];
  const float* out_b     = (const float*)d_in[38];

  float* outF = (float*)d_out;
  float* outPred = outF;                 // (16,1000,16)
  float* outCert = outF + 256000;        // (16,2,16)
  float* outSos  = outF + 256512;        // (16,8256)

  // ---- workspace carve-out (256B aligned) ----
  size_t off = 0;
  char* base = (char*)d_ws;
  auto A8 = [&](size_t bytes) -> void* {
    off = (off + 255) & ~(size_t)255;
    void* p = base + off; off += bytes; return p;
  };
  unsigned short* W2T  = (unsigned short*)A8((size_t)4608 * 512 * 2);
  unsigned short* kvT  = (unsigned short*)A8((size_t)512 * 512 * 2);
  unsigned short* WqhT = (unsigned short*)A8((size_t)512 * 512 * 2);
  unsigned short* WkT  = (unsigned short*)A8((size_t)512 * 512 * 2);
  unsigned short* WvT  = (unsigned short*)A8((size_t)512 * 512 * 2);
  unsigned short* WoT  = (unsigned short*)A8((size_t)512 * 512 * 2);
  unsigned short* qT   = (unsigned short*)A8((size_t)REPAz * 512 * 2);
  unsigned short* synT = (unsigned short*)A8((size_t)2560 * 4096 * 2);
  unsigned short* outT = (unsigned short*)A8((size_t)REPOz * 1008 * 2);
  unsigned short* W1fT = (unsigned short*)A8((size_t)KNLM * 2048 * 2);
  unsigned short* h1p  = (unsigned short*)A8((size_t)16 * 34 * 34 * 512 * 2);
  float* conv2raw = (float*)A8((size_t)16 * 1024 * 512 * 4);
  float* feat  = (float*)A8((size_t)4096 * 512 * 4);
  float* kvbuf = (float*)A8((size_t)4096 * 512 * 4);
  float* Kbuf  = (float*)A8((size_t)4096 * 512 * 4);
  float* Vbuf  = (float*)A8((size_t)4096 * 512 * 4);
  float* aA = (float*)A8((size_t)16 * REPAz * 4);
  float* bA = (float*)A8((size_t)16 * REPAz * 4);
  float* sa = (float*)A8((size_t)16 * REPAz * 4);
  float* aO = (float*)A8((size_t)16 * REPOz * 4);
  float* bO = (float*)A8((size_t)16 * REPOz * 4);
  float* so = (float*)A8((size_t)16 * REPOz * 4);
  unsigned short* IA = (unsigned short*)A8((size_t)REPAz * 2);
  unsigned short* JA = (unsigned short*)A8((size_t)REPAz * 2);
  unsigned short* IO = (unsigned short*)A8((size_t)REPOz * 2);
  unsigned short* JO = (unsigned short*)A8((size_t)REPOz * 2);
  float* rA = (float*)A8((size_t)REPAz * 4);
  float* rO = (float*)A8((size_t)REPOz * 4);
  float* hist  = (float*)A8((size_t)16 * 2048 * 16 * 4);
  float* post  = (float*)A8((size_t)16 * 2048 * 4);
  float* qC    = (float*)A8((size_t)16 * 512 * 4);
  float* qln   = (float*)A8((size_t)16 * 512 * 4);
  float* qh    = (float*)A8((size_t)16 * 512 * 4);
  float* attO  = (float*)A8((size_t)16 * 512 * 4);
  float* oproj = (float*)A8((size_t)16 * 512 * 4);
  float* zin   = (float*)A8((size_t)16 * 2560 * 4);
  float* zbuf  = (float*)A8((size_t)16 * 4096 * 4);
  float* preraw= (float*)A8((size_t)16 * 2048 * 4);
  float* pre   = (float*)A8((size_t)16 * 2048 * 4);
  float* outC  = (float*)A8((size_t)16 * 1008 * 4);
  float* c_h   = (float*)A8((size_t)2048 * 4);

  // ================= setup phase =================
  // weight conversion/transposition to bf16 (read fp32 once, stream bf16 16x)
  k_tr<<<dim3(16, 16), 256, 0, stream>>>(kvT,  kv_w,       512, 512, 512, 0);
  k_tr<<<dim3(16, 16), 256, 0, stream>>>(WqhT, attn_in_w,  512, 512, 512, 0);
  k_tr<<<dim3(16, 16), 256, 0, stream>>>(WkT,  attn_in_w,  512, 512, 512, 512);
  k_tr<<<dim3(16, 16), 256, 0, stream>>>(WvT,  attn_in_w,  512, 512, 512, 1024);
  k_tr<<<dim3(16, 16), 256, 0, stream>>>(WoT,  attn_out_w, 512, 512, 512, 0);
  k_tr<<<dim3(1028, 16), 256, 0, stream>>>(qT,   q_w,   512,  REPAz, 512, 0);
  k_tr<<<dim3(80, 128), 256, 0, stream>>>(synT, syn_w, 4096, 2560, 4096, 0);
  k_tr<<<dim3(258, 32), 256, 0, stream>>>(outT, out_w, 1000, REPOz, 1008, 0);
  k_conv2wt<<<cdiv(4608 * 512, 256), 256, 0, stream>>>(W2T, conv2_w);
  k_nlmfuse<<<dim3(64, 64, 16), 256, 0, stream>>>(W1fT, nlm_w1, nlm_w2);
  k_nlmbias<<<2048, 256, 0, stream>>>(nlm_b1, nlm_w2, nlm_b2, c_h);
  k_pairidx<<<cdiv(REPAz, 256), 256, 0, stream>>>(IA, JA, rA, decay_a, NSAz, REPAz);
  k_pairidx<<<cdiv(REPOz, 256), 256, 0, stream>>>(IO, JO, rO, decay_o, NSOz, REPOz);

  // recurrent state init (deterministic every call)
  k_fzero<<<cdiv(16 * REPAz, 256), 256, 0, stream>>>(aA, 16 * REPAz);
  k_fzero<<<cdiv(16 * REPAz, 256), 256, 0, stream>>>(bA, 16 * REPAz);
  k_fzero<<<cdiv(16 * REPOz, 256), 256, 0, stream>>>(aO, 16 * REPOz);
  k_fzero<<<cdiv(16 * REPOz, 256), 256, 0, stream>>>(bO, 16 * REPOz);
  k_histinit<<<cdiv(16 * 2048 * 16, 256), 256, 0, stream>>>(hist, hist_init);
  k_postinit<<<cdiv(16 * 2048, 256), 256, 0, stream>>>(post, post_init);

  // ---- frontend ----
  k_hzero<<<cdiv(16 * 34 * 34 * 512, 256), 256, 0, stream>>>(h1p, 16 * 34 * 34 * 512);
  k_conv1<<<cdiv(16 * 32 * 32 * 512, 256), 256, 0, stream>>>(
      in_x, conv1_w, conv1_b, bn1_g, bn1_b, bn1_m, bn1_v, h1p);
  k_rowbias<<<cdiv(16384 * 512, 256), 256, 0, stream>>>(conv2raw, conv2_b, 16384, 512, 512, 512);
  k_conv2gemm<8><<<dim3(4, 1024, 9), 32, 0, stream>>>(h1p, W2T, conv2raw, 512);
  k_pool2feat<<<cdiv(16 * 256 * 512, 256), 256, 0, stream>>>(
      conv2raw, bn2_g, bn2_b, bn2_m, bn2_v, feat);

  // kv = LN(feat @ kv_w^T + b); K,V projections (scan-invariant, store mode)
  k_gemm<8><<<dim3(4, 256, 1), 32, 0, stream>>>(feat, 512, kvT, 512, kvbuf, 512, 512, 512, kv_b);
  k_layernorm<<<4096, 256, 0, stream>>>(kvbuf, kvbuf, kv_ln_g, kv_ln_b, 512);
  k_gemm<8><<<dim3(4, 256, 1), 32, 0, stream>>>(kvbuf, 512, WkT, 512, Kbuf, 512, 512, 512, attn_in_b + 512);
  k_gemm<8><<<dim3(4, 256, 1), 32, 0, stream>>>(kvbuf, 512, WvT, 512, Vbuf, 512, 512, 512, attn_in_b + 1024);

  // ================= recurrent scan (M=16 steps) =================
  for (int m = 0; m < Mz; ++m) {
    // action pair synapses + decay -> sa (A of q GEMM)
    k_pair<<<cdiv(16 * REPAz, 256), 256, 0, stream>>>(
        post, Dz - NSAz, IA, JA, rA, aA, bA, sa, REPAz);
    // q = LN(sa @ q_w^T + q_b)   [16 x 32896 x 512, HBM-bound bf16 stream]
    k_rowbias<<<32, 256, 0, stream>>>(qC, q_b, 16, 512, 512, 512);
    k_gemm<8><<<dim3(4, 1, 80), 32, 0, stream>>>(sa, REPAz, qT, 512, qC, 512, REPAz, 416, nullptr);
    k_layernorm<<<16, 256, 0, stream>>>(qC, qln, q_ln_g, q_ln_b, 512);
    // qh = q @ wq^T + bq   (store mode)
    k_gemm<8><<<dim3(4, 1, 1), 32, 0, stream>>>(qln, 512, WqhT, 512, qh, 512, 512, 512, attn_in_b);
    // attention + output projection (store mode)
    k_attn<<<128, 256, 0, stream>>>(qh, Kbuf, Vbuf, attO);
    k_gemm<8><<<dim3(4, 1, 1), 32, 0, stream>>>(attO, 512, WoT, 512, oproj, 512, 512, 512, attn_out_b);
    // synapse GLU: z = [o, post] @ syn_w^T + b ; pre = LN(z[:D]*sigmoid(z[D:]))
    k_zin<<<cdiv(16 * 2560, 256), 256, 0, stream>>>(zin, oproj, post);
    k_rowbias<<<cdiv(16 * 4096, 256), 256, 0, stream>>>(zbuf, syn_b, 16, 4096, 4096, 4096);
    k_gemm<8><<<dim3(32, 1, 8), 32, 0, stream>>>(zin, 2560, synT, 4096, zbuf, 4096, 2560, 320, nullptr);
    k_glu<<<cdiv(16 * 2048, 256), 256, 0, stream>>>(preraw, zbuf);
    k_layernorm<<<16, 256, 0, stream>>>(preraw, pre, syn_ln_g, syn_ln_b, 2048);
    // history window + neuron-level model (fused bf16 weight, dominant stream)
    k_histshift<<<cdiv(16 * 2048, 256), 256, 0, stream>>>(hist, pre);
    k_rowbias<<<cdiv(16 * 2048, 256), 256, 0, stream>>>(post, c_h, 16, 2048, 2048, 2048);
    k_gemm<8><<<dim3(16, 1, 64), 32, 0, stream>>>(hist, KNLM, W1fT, 2048, post, 2048, KNLM, 512, nullptr);
    // output pair synapses + decay -> so
    k_pair<<<cdiv(16 * REPOz, 256), 256, 0, stream>>>(
        post, 0, IO, JO, rO, aO, bO, so, REPOz);
    // pred = so @ out_w^T + out_b (N padded to 1008)
    k_rowbias<<<cdiv(16 * 1008, 256), 256, 0, stream>>>(outC, out_b, 16, 1008, 1008, 1000);
    k_gemm<7><<<dim3(9, 1, 20), 32, 0, stream>>>(so, REPOz, outT, 1008, outC, 1008, REPOz, 416, nullptr);
    // write predictions[:, :, m] and certainties[:, :, m]
    k_softout<<<16, 256, 0, stream>>>(outC, outPred, outCert, m);
  }
  // sos = so from last step
  k_copy<<<cdiv(16 * REPOz, 256), 256, 0, stream>>>(outSos, so, 16 * REPOz);
}
